// ReactorNet_43018392437095
// MI455X (gfx1250) — compile-verified, matrix operand order checked by reading, not run on hardware
//
#include <hip/hip_runtime.h>

// ---------------------------------------------------------------------------
// Constant-pressure adiabatic reactor batch, backward Euler + Newton + LU.
// One wave32 per reactor. Jacobian species-species block computed as
//   M = nu_net^T * diag(rate) * nu_react     (53x53, K=325)
// with v_wmma_f32_16x16x32_bf16, rate split into hi+lo bf16 for ~fp32 accuracy.
// Sparse stoichiometry tables (2 reactants / <=4 net entries) staged in LDS
// drive the Arrhenius rates and wdot scatter (ds_add_f32 atomics).
// ---------------------------------------------------------------------------

#define NBATCH  1024
#define NS      53
#define NR      325
#define NST     54      // state size (T + Y)
#define NP      64      // padded system size
#define RP      352     // K padded to multiple of 32
#define KT      11      // RP / 32
#define DTS     1e-7f
#define RGAS    8.314462618f
#define LDJ     65      // LDS row stride for J (conflict-free columns)
#define NEWTON  3
#define NSTEPS  16

typedef __attribute__((ext_vector_type(16))) __bf16 bf16x16;
typedef __attribute__((ext_vector_type(8)))  float  f32x8;
typedef unsigned int u32x4 __attribute__((ext_vector_type(4)));

union Frag { u32x4 q[2]; bf16x16 v; };

struct Gp {
  const float *T0, *Y0, *W, *beta, *EaR, *nasa, *Pp;
  const float *lnA;                  // ws: precomputed log(A), padded to RP
  const unsigned int *rxn;           // ws: packed sparse tables, 3 words/rxn
  const __bf16 *nuAT, *nuBT;         // ws: bf16 transposed nu operands
};

struct SharedBlk {
  float J[NP * LDJ];
  float S[NP], Pv[NP], F[NP], G[NP];
  float invC[NP], logC[NP], cmask[NP];
  float H[NP], cpR[NP], W[NP], mask[NP];
  float wdot[NP], wT[NP], SM[NP], HM[NP];
  float rate[RP];
  float beta[RP], EaR[RP], lnA[RP];
  unsigned int rx0[RP], rx1[RP], rx2[RP];
  float nasa[372];                   // 53*7 padded
};

struct Thermo {
  float T, lnT, rho, Wbar, cp, cpp, fT, CPW, HwT, tmask;
};

__device__ __forceinline__ float waveSum(float v) {
#pragma unroll
  for (int m = 16; m >= 1; m >>= 1) v += __shfl_xor(v, m, 32);
  return v;
}
__device__ __forceinline__ float halfSum(float v) {   // sum within 16-lane halves
#pragma unroll
  for (int m = 8; m >= 1; m >>= 1) v += __shfl_xor(v, m, 32);
  return v;
}

// ---------------------------------------------------------------------------
// Setup 1: padded transposed nu matrices (bf16 WMMA operands).
// nuAT[i][r] = nu_net[r][i]  (A operand),  nuBT[n][r] = nu_react[r][n].
// ---------------------------------------------------------------------------
__global__ void setup_nu(const float* __restrict__ nuR, const float* __restrict__ nuP,
                         __bf16* __restrict__ nuAT, __bf16* __restrict__ nuBT) {
  int idx = blockIdx.x * blockDim.x + threadIdx.x;
  if (idx >= NP * RP) return;
  int i = idx / RP, r = idx % RP;
  float a = 0.f, b = 0.f;
  if (i < NS && r < NR) { b = nuR[r * NS + i]; a = nuP[r * NS + i] - b; }
  nuAT[idx] = (__bf16)a;
  nuBT[idx] = (__bf16)b;
}

// ---------------------------------------------------------------------------
// Setup 2: sparse stoichiometry tables + log(A).
//  word0: reactant i0 | i1<<8 | count<<16     (dot = logC[i0] (+ logC[i1]))
//  word1: 4 x net species index (8b each)
//  word2: 4 x net sign encoded (sign+1): 0 -> -1, 1 -> none, 2 -> +1
// ---------------------------------------------------------------------------
__global__ void setup_sparse(const float* __restrict__ nuR, const float* __restrict__ nuP,
                             const float* __restrict__ A,
                             float* __restrict__ lnA, unsigned int* __restrict__ rxn) {
  int r = blockIdx.x * blockDim.x + threadIdx.x;
  if (r >= RP) return;
  if (r < NR) {
    lnA[r] = __logf(A[r]);
    int ri[2]; int rn = 0;
    int ni[4]; int sg[4]; int nn = 0;
    for (int s = 0; s < NS; ++s) {
      float vr = nuR[r * NS + s];
      float vnet = nuP[r * NS + s] - vr;
      if (vr > 0.5f && rn < 2) ri[rn++] = s;
      if ((vnet > 0.5f || vnet < -0.5f) && nn < 4) {
        ni[nn] = s; sg[nn] = (vnet > 0.f) ? 1 : -1; ++nn;
      }
    }
    if (rn == 0) { ri[0] = 0; rn = 1; }
    if (rn == 1) ri[1] = ri[0];
    unsigned int w0 = (unsigned int)ri[0] | ((unsigned int)ri[1] << 8)
                    | ((unsigned int)rn << 16);
    unsigned int w1 = 0, w2 = 0;
    for (int q = 0; q < 4; ++q) {
      int idx = (q < nn) ? ni[q] : 0;
      int sv  = (q < nn) ? (sg[q] + 1) : 1;
      w1 |= ((unsigned int)idx) << (8 * q);
      w2 |= ((unsigned int)sv)  << (8 * q);
    }
    rxn[r * 3 + 0] = w0; rxn[r * 3 + 1] = w1; rxn[r * 3 + 2] = w2;
  } else {
    lnA[r] = 0.f;
    rxn[r * 3 + 0] = 0;
    rxn[r * 3 + 1] = 0;
    rxn[r * 3 + 2] = 0x01010101u;    // all "no entry"
  }
}

// ---------------------------------------------------------------------------
// rhs evaluation: per-species arrays + rates (sparse) + F, thermo scalars out
// ---------------------------------------------------------------------------
__device__ void eval_rhs(SharedBlk& sh, const float* st, float Pval,
                         int lane, Thermo& th) {
  float Traw = st[0];
  float T = fminf(fmaxf(Traw, 300.f), 3500.f);
  th.tmask = (Traw > 300.f && Traw < 3500.f) ? 1.f : 0.f;
  th.T = T;
  th.lnT = __logf(T);

  float ycv[2];
  float invWp = 0.f;
#pragma unroll
  for (int q = 0; q < 2; ++q) {
    int j = lane + 32 * q;
    float yc = 0.f, mk = 0.f;
    if (j < NS) {
      float y = st[1 + j];
      yc = fmaxf(y, 0.f);
      mk = (y > 0.f) ? 1.f : 0.f;
      invWp += yc / sh.W[j];
    }
    ycv[q] = yc;
    sh.mask[j] = mk;
    sh.wdot[j] = 0.f;            // zero scatter accumulators
    sh.wT[j]   = 0.f;
  }
  float invW = waveSum(invWp);
  th.Wbar = 1.f / invW;
  th.rho = Pval * th.Wbar / (RGAS * T);

  float T2 = T * T, T3 = T2 * T, T4 = T2 * T2;
  float cp_p = 0.f, cpp_p = 0.f;
#pragma unroll
  for (int q = 0; q < 2; ++q) {
    int j = lane + 32 * q;
    float C = 0.f, h = 0.f, cpr = 0.f;
    if (j < NS) {
      float yc = ycv[q];
      float Wj = sh.W[j];
      C = th.rho * yc / Wj;
      const float* a = sh.nasa + j * 7;
      cpr = a[0] + a[1]*T + a[2]*T2 + a[3]*T3 + a[4]*T4;
      float cprp = a[1] + 2.f*a[2]*T + 3.f*a[3]*T2 + 4.f*a[4]*T3;
      float hRT = a[0] + 0.5f*a[1]*T + (1.f/3.f)*a[2]*T2 + 0.25f*a[3]*T3
                + 0.2f*a[4]*T4 + a[5] / T;
      h = hRT * RGAS * T;
      cp_p  += yc * cpr  * RGAS / Wj;
      cpp_p += yc * cprp * RGAS / Wj;
    }
    float Cc = fmaxf(C, 1e-30f);
    sh.logC[j] = __logf(Cc);
    bool on = (C > 1e-30f);
    sh.invC[j]  = on ? 1.f / C : 0.f;
    sh.cmask[j] = on ? 1.f : 0.f;
    sh.H[j] = h;
    sh.cpR[j] = cpr;
  }
  th.cp  = waveSum(cp_p);
  th.cpp = waveSum(cpp_p);
  __syncthreads();

  // Arrhenius + sparse law of mass action + scatter of wdot / wT
  float invT = 1.f / T;
  for (int r = lane; r < NR; r += 32) {
    unsigned int w0 = sh.rx0[r], w1 = sh.rx1[r], w2 = sh.rx2[r];
    int i0 = (int)(w0 & 255u);
    int i1 = (int)((w0 >> 8) & 255u);
    int rn = (int)((w0 >> 16) & 255u);
    float dot = sh.logC[i0];
    if (rn == 2) dot += sh.logC[i1];
    float be = sh.beta[r], ea = sh.EaR[r];
    float rate = __expf(sh.lnA[r] + be * th.lnT - ea * invT + dot);
    sh.rate[r] = rate;
    float ratT = rate * (be + ea * invT) * invT;
#pragma unroll
    for (int q2 = 0; q2 < 4; ++q2) {
      int idx = (int)((w1 >> (8 * q2)) & 255u);
      float cf = (float)((int)((w2 >> (8 * q2)) & 255u) - 1);
      if (cf != 0.f) {
        atomicAdd(&sh.wdot[idx], cf * rate);
        atomicAdd(&sh.wT[idx],   cf * ratT);
      }
    }
  }
  for (int r = NR + lane; r < RP; r += 32) sh.rate[r] = 0.f;
  __syncthreads();

  // energy sums + F
  float Qp = 0.f, CPWp = 0.f, HwTp = 0.f;
#pragma unroll
  for (int q = 0; q < 2; ++q) {
    int j = lane + 32 * q;
    if (j < NS) {
      float wd = sh.wdot[j], wt = sh.wT[j];
      Qp   += sh.H[j] * wd;
      CPWp += sh.cpR[j] * RGAS * wd;
      HwTp += sh.H[j] * wt;
    }
  }
  float Q = waveSum(Qp);
  th.CPW = waveSum(CPWp);
  th.HwT = waveSum(HwTp);
  th.fT  = -Q / (th.rho * th.cp);

#pragma unroll
  for (int q = 0; q < 2; ++q) {
    int idx = lane + 32 * q;
    float fv = 0.f;
    if (idx == 0) fv = th.fT;
    else if (idx <= NS) fv = sh.wdot[idx - 1] * sh.W[idx - 1] / th.rho;
    sh.F[idx] = fv;
  }
  __syncthreads();
}

// ---------------------------------------------------------------------------
// Jacobian: WMMA GEMM  M = nu_net^T diag(rate) nu_react  (split-bf16 rate),
// then chain rule into J = I - DT*df/ds (64x64 padded, LDS).
// ---------------------------------------------------------------------------
__device__ void build_jac(SharedBlk& sh, const Gp& gp, int lane, const Thermo& th) {
  const int nlo = lane & 15;
  const int hi  = lane >> 4;
  const int kb0 = hi * 8;

  f32x8 acc[4][4];
#pragma unroll
  for (int a = 0; a < 4; ++a)
#pragma unroll
    for (int b = 0; b < 4; ++b)
#pragma unroll
      for (int e = 0; e < 8; ++e) acc[a][b][e] = 0.f;

  float colMask[4];
#pragma unroll
  for (int nt = 0; nt < 4; ++nt) colMask[nt] = sh.cmask[nt * 16 + nlo];

  for (int kt = 0; kt < KT; ++kt) {
    const int kbase = kt * 32;
    // A fragments (16x32 bf16): two aligned 16B loads per tile per lane
    bf16x16 afrag[4];
#pragma unroll
    for (int mt = 0; mt < 4; ++mt) {
      const __bf16* pa = gp.nuAT + (mt * 16 + nlo) * RP + kbase + kb0;
      Frag u;
      u.q[0] = *(const u32x4*)(pa);
      u.q[1] = *(const u32x4*)(pa + 16);
      afrag[mt] = u.v;
    }
    // rate scale, split hi+lo in bf16 for near-fp32 product accuracy
    float rhiF[16], rloF[16];
#pragma unroll
    for (int e = 0; e < 16; ++e) {
      int K = kbase + ((e < 8) ? (kb0 + e) : (16 + kb0 + e - 8));
      float rv = sh.rate[K];
      __bf16 hb = (__bf16)rv;
      float hf = (float)hb;
      rhiF[e] = hf;
      rloF[e] = rv - hf;
    }
#pragma unroll
    for (int nt = 0; nt < 4; ++nt) {
      const __bf16* pb = gp.nuBT + (nt * 16 + nlo) * RP + kbase + kb0;
      Frag ub;
      ub.q[0] = *(const u32x4*)(pb);
      ub.q[1] = *(const u32x4*)(pb + 16);
      bf16x16 bH, bL;
      const float cm = colMask[nt];
#pragma unroll
      for (int e = 0; e < 16; ++e) {
        float nv = (float)ub.v[e] * cm;
        bH[e] = (__bf16)(nv * rhiF[e]);
        bL[e] = (__bf16)(nv * rloF[e]);
      }
#pragma unroll
      for (int mt = 0; mt < 4; ++mt) {
        acc[mt][nt] = __builtin_amdgcn_wmma_f32_16x16x32_bf16(
            false, afrag[mt], false, bH, (short)0, acc[mt][nt], false, false);
        acc[mt][nt] = __builtin_amdgcn_wmma_f32_16x16x32_bf16(
            false, afrag[mt], false, bL, (short)0, acc[mt][nt], false, false);
      }
    }
  }

  // Row sums SM_i (C/D layout: row = v + 8*hi + 16*mt, col = nt*16 + nlo)
  float smv[4][8];
#pragma unroll
  for (int mt = 0; mt < 4; ++mt)
#pragma unroll
    for (int v = 0; v < 8; ++v) {
      float t = acc[mt][0][v] + acc[mt][1][v] + acc[mt][2][v] + acc[mt][3][v];
      t = halfSum(t);
      smv[mt][v] = t;
      if (nlo == 0) sh.SM[mt * 16 + v + 8 * hi] = t;
    }
  // h-weighted column sums HM_k
#pragma unroll
  for (int nt = 0; nt < 4; ++nt) {
    float t = 0.f;
#pragma unroll
    for (int mt = 0; mt < 4; ++mt)
#pragma unroll
      for (int v = 0; v < 8; ++v)
        t = fmaf(sh.H[mt * 16 + v + 8 * hi], acc[mt][nt][v], t);
    t += __shfl_xor(t, 16, 32);
    if (hi == 0) sh.HM[nt * 16 + nlo] = t;
  }
  __syncthreads();
  float shp = sh.H[lane] * sh.SM[lane] + sh.H[lane + 32] * sh.SM[lane + 32];
  const float SHM = waveSum(shp);

  // J = identity (padding rows/cols stay identity)
  for (int idx = lane; idx < NP * NP; idx += 32) {
    int r = idx >> 6, c = idx & 63;
    sh.J[r * LDJ + c] = (r == c) ? 1.f : 0.f;
  }
  __syncthreads();

  // Interior: d f_Yi / d Y_k
  const float rhoinv = 1.f / th.rho;
#pragma unroll
  for (int mt = 0; mt < 4; ++mt)
#pragma unroll
    for (int v = 0; v < 8; ++v) {
      const int i = mt * 16 + v + 8 * hi;
      if (i < NS) {
        const float Wi = sh.W[i];
        const float r1 = Wi * th.Wbar * rhoinv * (sh.wdot[i] - smv[mt][v]);
#pragma unroll
        for (int nt = 0; nt < 4; ++nt) {
          const int k = nt * 16 + nlo;
          if (k < NS) {
            const float Wk = sh.W[k];
            float dF = (Wi / Wk) * acc[mt][nt][v] * sh.invC[k] + r1 / Wk;
            dF *= sh.mask[k];
            sh.J[(1 + i) * LDJ + (1 + k)] = ((i == k) ? 1.f : 0.f) - DTS * dF;
          }
        }
      }
    }

  // Borders: column 0 (d/dT) and row 0 (df_T/dx)
  const float invT = 1.f / th.T;
  const float rc = 1.f / (th.rho * th.cp);
#pragma unroll
  for (int q = 0; q < 2; ++q) {
    const int j = lane + 32 * q;
    if (j < NS) {
      const float Wj = sh.W[j];
      float dfdT = (Wj * rhoinv) * (sh.wT[j] + (sh.wdot[j] - sh.SM[j]) * invT);
      sh.J[(1 + j) * LDJ + 0] = -DTS * dfdT * th.tmask;
      float HY  = sh.HM[j] * sh.invC[j] * th.rho / Wj - SHM * th.Wbar / Wj;
      float dfT = -HY * rc - th.fT * (-th.Wbar / Wj + sh.cpR[j] * RGAS / (Wj * th.cp));
      sh.J[0 * LDJ + (1 + j)] = -DTS * dfT * sh.mask[j];
    }
  }
  if (lane == 0) {
    float dQdT  = th.CPW - SHM * invT + th.HwT;
    float dfTdT = -dQdT * rc - th.fT * (-invT + th.cpp / th.cp);
    sh.J[0] = 1.f - DTS * dfTdT * th.tmask;
  }
  __syncthreads();
}

// ---------------------------------------------------------------------------
// Main kernel: one wave32 per reactor
// ---------------------------------------------------------------------------
__global__ void __launch_bounds__(32)
reactor_kernel(Gp gp, float* __restrict__ out) {
  __shared__ SharedBlk sh;
  const int b = blockIdx.x;
  const int lane = threadIdx.x;
  const float Pval = gp.Pp[0];

  // Warm caches for the WMMA operand matrices (global_prefetch_b8)
  for (int off = lane * 128; off < NP * RP; off += 32 * 128) {
    __builtin_prefetch(gp.nuAT + off, 0, 0);
    __builtin_prefetch(gp.nuBT + off, 0, 0);
  }

  // Stage read-only per-reaction / per-species tables into LDS once
  for (int t = lane; t < RP; t += 32) {
    bool v = (t < NR);
    sh.beta[t] = v ? gp.beta[t] : 0.f;
    sh.EaR[t]  = v ? gp.EaR[t]  : 1.f;
    sh.lnA[t]  = gp.lnA[t];
    sh.rx0[t]  = gp.rxn[t * 3 + 0];
    sh.rx1[t]  = gp.rxn[t * 3 + 1];
    sh.rx2[t]  = gp.rxn[t * 3 + 2];
  }
  for (int t = lane; t < NS * 7; t += 32) sh.nasa[t] = gp.nasa[t];

#pragma unroll
  for (int q = 0; q < 2; ++q) {
    int j = lane + 32 * q;
    sh.W[j] = (j < NS) ? gp.W[j] : 1.f;
    float sv = 0.f;
    if (j == 0) sv = gp.T0[b];
    else if (j <= NS) sv = gp.Y0[b * NS + (j - 1)];
    sh.S[j] = sv;
  }
  __syncthreads();

  Thermo th;
  for (int step = 0; step < NSTEPS; ++step) {
#pragma unroll
    for (int q = 0; q < 2; ++q) { int j = lane + 32 * q; sh.Pv[j] = sh.S[j]; }
    __syncthreads();

    // explicit-Euler predictor
    eval_rhs(sh, sh.Pv, Pval, lane, th);
#pragma unroll
    for (int q = 0; q < 2; ++q) { int j = lane + 32 * q; sh.S[j] = sh.Pv[j] + DTS * sh.F[j]; }
    __syncthreads();

    for (int it = 0; it < NEWTON; ++it) {
      eval_rhs(sh, sh.S, Pval, lane, th);
      build_jac(sh, gp, lane, th);
#pragma unroll
      for (int q = 0; q < 2; ++q) {
        int j = lane + 32 * q;
        sh.G[j] = sh.S[j] - sh.Pv[j] - DTS * sh.F[j];
      }
      __syncthreads();

      // LU (no pivoting; J ~ I - DT*df, diag ~ 1), 2 rows per lane,
      // broadcast pivot element loaded once per column for both rows.
      for (int k = 0; k < NP - 1; ++k) {
        float pinv = 1.f / sh.J[k * LDJ + k];
        const int r0 = lane, r1 = lane + 32;
        const bool a0 = (r0 > k);
        float l0 = 0.f, l1 = 0.f;
        if (a0) { l0 = sh.J[r0 * LDJ + k] * pinv; sh.J[r0 * LDJ + k] = l0; }
        if (r1 > k) { l1 = sh.J[r1 * LDJ + k] * pinv; sh.J[r1 * LDJ + k] = l1; }
        for (int c = k + 1; c < NP; ++c) {
          float pv = sh.J[k * LDJ + c];
          if (a0) sh.J[r0 * LDJ + c] = fmaf(-l0, pv, sh.J[r0 * LDJ + c]);
          sh.J[r1 * LDJ + c] = fmaf(-l1, pv, sh.J[r1 * LDJ + c]);
        }
        __syncthreads();
      }
      // forward substitution (unit L)
      for (int k = 0; k < NP - 1; ++k) {
        float yk = sh.G[k];
#pragma unroll
        for (int q = 0; q < 2; ++q) {
          int r = lane + 32 * q;
          if (r > k) sh.G[r] = fmaf(-sh.J[r * LDJ + k], yk, sh.G[r]);
        }
        __syncthreads();
      }
      // backward substitution
      for (int k = NP - 1; k >= 0; --k) {
        float xk = sh.G[k] / sh.J[k * LDJ + k];
        __syncthreads();
        if (lane == 0) sh.G[k] = xk;
#pragma unroll
        for (int q = 0; q < 2; ++q) {
          int r = lane + 32 * q;
          if (r < k) sh.G[r] = fmaf(-sh.J[r * LDJ + k], xk, sh.G[r]);
        }
        __syncthreads();
      }
#pragma unroll
      for (int q = 0; q < 2; ++q) { int j = lane + 32 * q; sh.S[j] -= sh.G[j]; }
      __syncthreads();
    }
  }

#pragma unroll
  for (int q = 0; q < 2; ++q) {
    int j = lane + 32 * q;
    if (j < NST) out[b * NST + j] = sh.S[j];
  }
}

// ---------------------------------------------------------------------------
extern "C" void kernel_launch(void* const* d_in, const int* in_sizes, int n_in,
                              void* d_out, int out_size, void* d_ws, size_t ws_size,
                              hipStream_t stream) {
  (void)in_sizes; (void)n_in; (void)out_size; (void)ws_size;
  const float* T0   = (const float*)d_in[0];
  const float* P    = (const float*)d_in[1];
  const float* Y0   = (const float*)d_in[2];
  const float* W    = (const float*)d_in[3];
  const float* A    = (const float*)d_in[4];
  const float* beta = (const float*)d_in[5];
  const float* EaR  = (const float*)d_in[6];
  const float* nuR  = (const float*)d_in[7];
  const float* nuP  = (const float*)d_in[8];
  const float* nasa = (const float*)d_in[9];

  char* ws = (char*)d_ws;
  __bf16* nuAT       = (__bf16*)ws;                              // 45056 B
  __bf16* nuBT       = (__bf16*)(ws + NP * RP * 2);              // 45056 B
  float*  lnA        = (float*)(ws + 2 * NP * RP * 2);           // 1408 B
  unsigned int* rxn  = (unsigned int*)(ws + 2 * NP * RP * 2 + RP * 4); // 4224 B

  setup_nu<<<(NP * RP + 255) / 256, 256, 0, stream>>>(nuR, nuP, nuAT, nuBT);
  setup_sparse<<<(RP + 127) / 128, 128, 0, stream>>>(nuR, nuP, A, lnA, rxn);

  Gp gp;
  gp.T0 = T0; gp.Y0 = Y0; gp.W = W; gp.beta = beta; gp.EaR = EaR;
  gp.nasa = nasa; gp.Pp = P;
  gp.lnA = lnA; gp.rxn = rxn; gp.nuAT = nuAT; gp.nuBT = nuBT;

  reactor_kernel<<<NBATCH, 32, 0, stream>>>(gp, (float*)d_out);
}
